// EFAdLIF_25400436588603
// MI455X (gfx1250) — matrix operand order repaired and checked by reading, not compiled, not run on hardware
//
#include <hip/hip_runtime.h>

// -------- CDNA5 (gfx1250) types --------
typedef __attribute__((ext_vector_type(16))) __bf16        v16bf;
typedef __attribute__((ext_vector_type(8)))  float         v8f;
typedef __attribute__((ext_vector_type(4)))  float         f32x4;
typedef __attribute__((ext_vector_type(4)))  unsigned int  u32x4;
typedef __attribute__((ext_vector_type(2)))  unsigned int  u32x2;
typedef __attribute__((ext_vector_type(8)))  int           v8i;
typedef __attribute__((ext_vector_type(4)))  int           v4i;

#define B_DIM   16384
#define IN_DIM  512
#define OUT_DIM 512
#define K_DIM   1024   // concatenated K: [input | z_tm1] vs [weight | recurrent]
#define BM      128    // block rows (batch)
#define BN      64     // block cols (neurons)
#define BK      64     // K per LDS stage (never straddles the 512 boundary)
#define LDA     72     // padded LDS row stride in shorts (16B-aligned, bank-spread)

// therock-10.0 headers ship the TDM header and the 6-arg builtin; ROCm 7.2 is 5-arg.
#if __has_include(<hip/amd_detail/amd_gfx1250_TDM.h>)
#define TDM_ARITY6 1
#endif

__device__ __forceinline__ unsigned int f32_to_bf16_rne(float f) {
  unsigned int u = __builtin_bit_cast(unsigned int, f);
  return (u + 0x7FFFu + ((u >> 16) & 1u)) >> 16;
}
__device__ __forceinline__ float bf16_to_f32(unsigned int h) {
  return __builtin_bit_cast(float, h << 16);
}

union FragU { v16bf v; u32x4 q[2]; };

// ---------------------------------------------------------------------------
// Pre-kernel: split [weight | recurrent] (512 x 1024 fp32, K-major) into
// bf16 hi + bf16 lo residual planes in workspace (2 MB total). Done ONCE per
// launch instead of once per row-block inside the GEMM (128x less conversion).
// ---------------------------------------------------------------------------
__global__ __launch_bounds__(256) void efadlif_wsplit(
    const float* __restrict__ w, const float* __restrict__ r,
    unsigned short* __restrict__ hi, unsigned short* __restrict__ lo)
{
  const size_t q = (size_t)blockIdx.x * 256 + threadIdx.x;  // quad index
  const size_t e = q * 4;                                   // element index
  const int n = (int)(e >> 10);          // neuron row
  const int k = (int)(e & 1023);         // concatenated K
  const float* src = (k < IN_DIM) ? (w + (size_t)n * IN_DIM + k)
                                  : (r + (size_t)n * IN_DIM + (k - IN_DIM));
  const f32x4 v = *(const f32x4*)src;
  unsigned int h[4], l[4];
#pragma unroll
  for (int e4 = 0; e4 < 4; ++e4) {
    h[e4] = f32_to_bf16_rne(v[e4]);
    l[e4] = f32_to_bf16_rne(v[e4] - bf16_to_f32(h[e4]));
  }
  u32x2 ph, pl;
  ph.x = h[0] | (h[1] << 16); ph.y = h[2] | (h[3] << 16);
  pl.x = l[0] | (l[1] << 16); pl.y = l[2] | (l[3] << 16);
  *(u32x2*)&hi[e] = ph;
  *(u32x2*)&lo[e] = pl;
}

// ---------------------------------------------------------------------------
// Tensor Data Mover: async-copy a 64(row) x 64(K) bf16 tile from the split
// weight plane (row stride 1024 elem) into LDS, with TDM padding configured
// to reproduce the LDA=72-short (144 B) bank-padded row stride.
// ---------------------------------------------------------------------------
__device__ __forceinline__ void tdm_load_tile(unsigned lds_off,
                                              const unsigned short* base,
                                              int row0, int kb)
{
  const unsigned long long ga =
      (unsigned long long)(uintptr_t)(base + (size_t)row0 * K_DIM + kb);
  u32x4 g0;
  g0.x = 1u;                                   // count=1 valid user descriptor
  g0.y = lds_off;                              // LDS byte address of tile
  g0.z = (unsigned)(ga & 0xFFFFFFFFu);         // global_addr[31:0]
  g0.w = (unsigned)((ga >> 32) & 0x1FFFFFFu)   // global_addr[56:32]
       | 0x80000000u;                          // type=2 ("image")
  v8i g1;
  g1[0] = (int)((1u << 16)     // data_size = 2 bytes (bf16)
              | (1u << 20)     // pad_enable
              | (4u << 22)     // pad_interval: 32 DWORDs = one 128 B tile row
              | (3u << 25));   // pad_amount: 4 DWORDs = 16 B -> 144 B stride
  g1[1] = (int)(((unsigned)K_DIM & 0xFFFFu) << 16);          // tensor_dim0 lo
  g1[2] = (int)((((unsigned)K_DIM >> 16) & 0xFFFFu)          // tensor_dim0 hi
              | (((unsigned)OUT_DIM & 0xFFFFu) << 16));      // tensor_dim1 lo
  g1[3] = (int)((((unsigned)OUT_DIM >> 16) & 0xFFFFu)        // tensor_dim1 hi
              | ((unsigned)BK << 16));                       // tile_dim0 = 64
  g1[4] = (int)(unsigned)BN;    // tile_dim1 = 64 rows, tile_dim2 = 0
  g1[5] = (int)K_DIM;           // tensor_dim0_stride[31:0] = 1024
  g1[6] = 0;                    // stride hi, dim1_stride lo (unused for 2D)
  g1[7] = 0;
  const v4i z4 = {0, 0, 0, 0};
#ifdef TDM_ARITY6
  const v8i z8 = {0, 0, 0, 0, 0, 0, 0, 0};
  __builtin_amdgcn_tensor_load_to_lds(g0, g1, z4, z4, z8, 0);
#else
  __builtin_amdgcn_tensor_load_to_lds(g0, g1, z4, z4, 0);
#endif
}

// ---------------------------------------------------------------------------
// One K-stage: TDM B tiles + convert/stage A tile + 2x K=32 WMMA sweeps.
// WITH_ALO is a compile-time flag (branch-free inner loops): the z_tm1 half
// of K has spikes in {0,1} which are exact in bf16, so its A-lo plane is 0.
// ---------------------------------------------------------------------------
template<bool WITH_ALO>
__device__ __forceinline__ void gemm_stage(
    const float* __restrict__ Xs, int kx, int row0, int col0, int kb,
    const unsigned short* __restrict__ wsH,
    const unsigned short* __restrict__ wsL,
    unsigned ldsBH, unsigned ldsBL,
    unsigned short (&AsH)[BM][LDA], unsigned short (&AsL)[BM][LDA],
    unsigned short (&BsH)[BN][LDA], unsigned short (&BsL)[BN][LDA],
    int t, int wave, int lm, int half, int waveM, int waveN,
    v8f (&acc)[2][2])
{
  __syncthreads();   // previous-stage LDS reads complete before overwrite

  // ---- B tiles: async TDM copies (one wave issues, EXEC-independent) ----
  if (wave == 0) {
    tdm_load_tile(ldsBH, wsH, col0, kb);
    tdm_load_tile(ldsBL, wsL, col0, kb);
  }

  // ---- A tile: 128 rows x 64 K, fp32 -> bf16 hi(/lo) ----
#pragma unroll
  for (int j = 0; j < 8; ++j) {
    const int idx = t + 256 * j;
    const int r = idx >> 4, c4 = idx & 15;
    const f32x4 v = *(const f32x4*)(Xs + (size_t)(row0 + r) * IN_DIM + kx + c4 * 4);
    unsigned int h[4];
#pragma unroll
    for (int e = 0; e < 4; ++e) h[e] = f32_to_bf16_rne(v[e]);
    u32x2 ph;
    ph.x = h[0] | (h[1] << 16); ph.y = h[2] | (h[3] << 16);
    *(u32x2*)&AsH[r][c4 * 4] = ph;
    if (WITH_ALO) {
      unsigned int l[4];
#pragma unroll
      for (int e = 0; e < 4; ++e)
        l[e] = f32_to_bf16_rne(v[e] - bf16_to_f32(h[e]));
      u32x2 pl;
      pl.x = l[0] | (l[1] << 16); pl.y = l[2] | (l[3] << 16);
      *(u32x2*)&AsL[r][c4 * 4] = pl;
    }
  }

  if (wave == 0) __builtin_amdgcn_s_wait_tensorcnt(0);
  __syncthreads();   // TDM + ds_store data visible to all waves

  // ---- WMMA over this K stage (two K=32 steps) ----
#pragma unroll
  for (int ks = 0; ks < BK; ks += 32) {
    FragU aH[2], aL[2], bH[2], bL[2];
#pragma unroll
    for (int ni = 0; ni < 2; ++ni) {
      const int br  = waveN * 32 + ni * 16 + lm;  // B lane = N column
      const int kb2 = ks + half * 16;             // lane-half: K 0..15 / 16..31
      bH[ni].q[0] = *(const u32x4*)&BsH[br][kb2];
      bH[ni].q[1] = *(const u32x4*)&BsH[br][kb2 + 8];
      bL[ni].q[0] = *(const u32x4*)&BsL[br][kb2];
      bL[ni].q[1] = *(const u32x4*)&BsL[br][kb2 + 8];
    }
#pragma unroll
    for (int mi = 0; mi < 2; ++mi) {
      const int ar = waveM * 32 + mi * 16 + lm;   // A lane = M row
      const int ka = ks + half * 16;              // K 0..15 / 16..31
      aH[mi].q[0] = *(const u32x4*)&AsH[ar][ka];
      aH[mi].q[1] = *(const u32x4*)&AsH[ar][ka + 8];
      if (WITH_ALO) {
        aL[mi].q[0] = *(const u32x4*)&AsL[ar][ka];
        aL[mi].q[1] = *(const u32x4*)&AsL[ar][ka + 8];
      }
    }
    // Term-major issue order: dependent WMMAs on the same accumulator are
    // 4 instructions apart (C->D chains overlap on the matrix pipe).
#pragma unroll
    for (int mi = 0; mi < 2; ++mi)
#pragma unroll
      for (int ni = 0; ni < 2; ++ni)
        acc[mi][ni] = __builtin_amdgcn_wmma_f32_16x16x32_bf16(
            false, aH[mi].v, false, bH[ni].v, (short)0, acc[mi][ni], false, false);
#pragma unroll
    for (int mi = 0; mi < 2; ++mi)
#pragma unroll
      for (int ni = 0; ni < 2; ++ni)
        acc[mi][ni] = __builtin_amdgcn_wmma_f32_16x16x32_bf16(
            false, aH[mi].v, false, bL[ni].v, (short)0, acc[mi][ni], false, false);
    if (WITH_ALO) {
#pragma unroll
      for (int mi = 0; mi < 2; ++mi)
#pragma unroll
        for (int ni = 0; ni < 2; ++ni)
          acc[mi][ni] = __builtin_amdgcn_wmma_f32_16x16x32_bf16(
              false, aL[mi].v, false, bH[ni].v, (short)0, acc[mi][ni], false, false);
    }
  }
}

// ---------------------------------------------------------------------------
// Fused GEMM + EFAdLIF neuron update.
// ---------------------------------------------------------------------------
__global__ __launch_bounds__(256) void efadlif_fused(
    const float* __restrict__ x,      const float* __restrict__ u_tm1,
    const float* __restrict__ z_tm1,  const float* __restrict__ w_tm1,
    const float* __restrict__ bias,   const float* __restrict__ a_p,
    const float* __restrict__ b_p,    const float* __restrict__ decay_u,
    const float* __restrict__ decay_w,
    const unsigned short* __restrict__ wsH,
    const unsigned short* __restrict__ wsL,
    float* __restrict__ z_out, float* __restrict__ u_out, float* __restrict__ w_out)
{
  __shared__ unsigned short AsH[BM][LDA];   // activations, bf16 hi
  __shared__ unsigned short AsL[BM][LDA];   // activations, bf16 lo residual
  __shared__ unsigned short BsH[BN][LDA];   // weights hi (TDM-filled)
  __shared__ unsigned short BsL[BN][LDA];   // weights lo (TDM-filled)

  const int t     = threadIdx.x;
  const int lane  = t & 31;
  const int wave  = t >> 5;
  const int lm    = lane & 15;   // N / M index within a 16-wide tile
  const int half  = lane >> 4;   // lane-half selects K sub-range (contiguous K 0..15 / 16..31)
  const int waveM = wave & 3;    // 4 wave-rows  -> 32 batch rows each
  const int waveN = wave >> 2;   // 2 wave-cols  -> 32 neuron cols each
  const int row0  = blockIdx.y * BM;
  const int col0  = blockIdx.x * BN;

  // Prefetch epilogue state into L2 while the GEMM runs (global_prefetch_b8).
  {
    const int pr = row0 + (t & 127);
    const int pc = col0 + ((t >> 7) << 5);
    __builtin_prefetch(&u_tm1[(size_t)pr * OUT_DIM + pc], 0, 1);
    __builtin_prefetch(&w_tm1[(size_t)pr * OUT_DIM + pc], 0, 1);
  }

  const unsigned ldsBH = (unsigned)(uintptr_t)&BsH[0][0];
  const unsigned ldsBL = (unsigned)(uintptr_t)&BsL[0][0];

  v8f acc[2][2];
#pragma unroll
  for (int mi = 0; mi < 2; ++mi)
#pragma unroll
    for (int ni = 0; ni < 2; ++ni)
      acc[mi][ni] = (v8f){0.f, 0.f, 0.f, 0.f, 0.f, 0.f, 0.f, 0.f};

  // K = [0, 512): input_tensor x weight — fp32 data needs the lo residual.
  for (int kk = 0; kk < IN_DIM / BK; ++kk)
    gemm_stage<true>(x, kk * BK, row0, col0, kk * BK, wsH, wsL,
                     ldsBH, ldsBL, AsH, AsL, BsH, BsL,
                     t, wave, lm, half, waveM, waveN, acc);

  // K = [512, 1024): z_tm1 x recurrent — spikes are exact in bf16 (no A-lo).
  for (int kk = 0; kk < OUT_DIM / BK; ++kk)
    gemm_stage<false>(z_tm1, kk * BK, row0, col0, IN_DIM + kk * BK, wsH, wsL,
                      ldsBH, ldsBL, AsH, AsL, BsH, BsL,
                      t, wave, lm, half, waveM, waveN, acc);

  // ---- Fused EFAdLIF epilogue (forward values; surrogate term is 0) ----
#pragma unroll
  for (int ni = 0; ni < 2; ++ni) {
    const int col = col0 + waveN * 32 + ni * 16 + lm;
    const float bv = bias[col];
    const float av = a_p[col];
    const float bb = b_p[col];
    const float du = decay_u[col];
    const float dw = decay_w[col];
#pragma unroll
    for (int mi = 0; mi < 2; ++mi) {
#pragma unroll
      for (int i = 0; i < 8; ++i) {
        // C/D 16x16 f32 layout: VGPR i, lanes0-15 -> M=i ; lanes16-31 -> M=i+8
        const int row = row0 + waveM * 32 + mi * 16 + i + half * 8;
        const size_t off = (size_t)row * OUT_DIM + col;
        const float soma = acc[mi][ni][i] + bv;
        const float up = __builtin_nontemporal_load(&u_tm1[off]);
        const float wp = __builtin_nontemporal_load(&w_tm1[off]);
        const float zp = z_tm1[off];
        const float ut = du * up + (1.0f - du) * (soma - wp);
        const float zt = (ut - 1.0f) > 0.0f ? 1.0f : 0.0f;   // heaviside(u-THR, 0)
        const float uo = ut * (1.0f - zt);                   // hard reset
        const float wt = dw * wp + (1.0f - dw) * (av * up + bb * zp) * 60.0f;
        __builtin_nontemporal_store(zt, &z_out[off]);
        __builtin_nontemporal_store(uo, &u_out[off]);
        __builtin_nontemporal_store(wt, &w_out[off]);
      }
    }
  }
}

extern "C" void kernel_launch(void* const* d_in, const int* in_sizes, int n_in,
                              void* d_out, int out_size, void* d_ws, size_t ws_size,
                              hipStream_t stream) {
  (void)in_sizes; (void)n_in; (void)out_size; (void)ws_size;
  const float* x    = (const float*)d_in[0];
  const float* u1   = (const float*)d_in[1];
  const float* z1   = (const float*)d_in[2];
  const float* w1   = (const float*)d_in[3];
  const float* wgt  = (const float*)d_in[4];
  const float* bia  = (const float*)d_in[5];
  const float* rec  = (const float*)d_in[6];
  const float* a    = (const float*)d_in[7];
  const float* b    = (const float*)d_in[8];
  const float* du   = (const float*)d_in[9];
  const float* dw   = (const float*)d_in[10];

  float* z_out = (float*)d_out;
  float* u_out = z_out + (size_t)B_DIM * OUT_DIM;
  float* w_out = u_out + (size_t)B_DIM * OUT_DIM;

  // Workspace: bf16 hi/lo planes of [weight | recurrent], 512x1024 each (2 MB).
  unsigned short* wsH = (unsigned short*)d_ws;
  unsigned short* wsL = wsH + (size_t)OUT_DIM * K_DIM;

  // 1) split weights once (512*1024/4 quads / 256 threads = 512 blocks)
  efadlif_wsplit<<<dim3((OUT_DIM * K_DIM) / (256 * 4)), 256, 0, stream>>>(
      wgt, rec, wsH, wsL);

  // 2) fused GEMM + neuron update (8 x 128 workgroups)
  dim3 grid(OUT_DIM / BN, B_DIM / BM);
  efadlif_fused<<<grid, 256, 0, stream>>>(x, u1, z1, w1, bia, a, b, du, dw,
                                          wsH, wsL, z_out, u_out, w_out);
}
